// AttentionLayer_53901839565224
// MI455X (gfx1250) — compile-verified, matrix operand order checked by reading, not run on hardware
//
#include <hip/hip_runtime.h>

typedef _Float16 half_t;
typedef __attribute__((ext_vector_type(16))) _Float16 v16h;
typedef __attribute__((ext_vector_type(8)))  float    v8f;

#define D_IN  2048
#define H_DIM 512
#define B_SZ  256
#define R_SZ  196
#define NREG  197   // R + 1

// pack two f32 -> packed f16 pair (one v_cvt_pkrtz_f16_f32), as raw 32 bits
__device__ __forceinline__ unsigned pk(float a, float b) {
    return __builtin_bit_cast(unsigned, __builtin_amdgcn_cvt_pkrtz(a, b));
}

// ---------------------------------------------------------------------------
// Generic fused GEMM: C[M,N] = act(A[M,K] @ W[K,N] + bias[N])
// act: 0=none, 1=relu, 2=tanh.  M%32==0, N%128==0, K%32==0.
// 256 threads (8 waves); block tile 32(M) x 128(N); K-step 32.
// Waves 2(M) x 4(N); each wave owns two 16x16x32 WMMA tiles.
// f32 -> f16 via v_cvt_pkrtz; A staged row-major (b64 stores), B staged
// column-major [n][k] stride 40 halves (16B-aligned frags, conflict-breaking).
// ---------------------------------------------------------------------------
__global__ __launch_bounds__(256)
void gemm_bias_act_kernel(const float* __restrict__ A, const float* __restrict__ W,
                          const float* __restrict__ bias, float* __restrict__ C,
                          int M, int N, int K, int act)
{
    __shared__ half_t la[32 * 32];    // [m][k]
    __shared__ half_t lb[128 * 40];   // [n][k] padded
    const int tid  = threadIdx.x;
    const int lane = tid & 31;
    const int wave = tid >> 5;
    const int wm   = wave >> 2;       // 0..1
    const int wn   = wave & 3;        // 0..3
    const int m_blk = blockIdx.y * 32;
    const int n_blk = blockIdx.x * 128;

    v8f acc0 = {}, acc1 = {};

    for (int k0 = 0; k0 < K; k0 += 32) {
        __syncthreads();
        // A tile 32x32: one float4/thread -> packed b64 LDS store
        {
            const int ml = tid >> 3;
            const int kl = (tid & 7) << 2;
            float4 f = *(const float4*)(A + (size_t)(m_blk + ml) * K + k0 + kl);
            *(uint2*)(&la[ml * 32 + kl]) = make_uint2(pk(f.x, f.y), pk(f.z, f.w));
        }
        // B tile 32x128 transposed: per pass thread handles k-pair x 4n
#pragma unroll
        for (int p = 0; p < 2; ++p) {
            const int k  = ((tid >> 5) + p * 8) << 1;   // 0,2,..,30
            const int n4 = (tid & 31) << 2;             // 0..124
            float4 f0 = *(const float4*)(W + (size_t)(k0 + k)     * N + n_blk + n4);
            float4 f1 = *(const float4*)(W + (size_t)(k0 + k + 1) * N + n_blk + n4);
            *(unsigned*)(&lb[(n4 + 0) * 40 + k]) = pk(f0.x, f1.x);
            *(unsigned*)(&lb[(n4 + 1) * 40 + k]) = pk(f0.y, f1.y);
            *(unsigned*)(&lb[(n4 + 2) * 40 + k]) = pk(f0.z, f1.z);
            *(unsigned*)(&lb[(n4 + 3) * 40 + k]) = pk(f0.w, f1.w);
        }
        __syncthreads();

        const int kh  = (lane >> 4) << 3;               // 0 or 8
        const int row = wm * 16 + (lane & 15);
        v16h a;
        {
            const half_t* p0 = &la[row * 32 + kh];
            const half_t* p1 = &la[row * 32 + 16 + kh];
#pragma unroll
            for (int i = 0; i < 8; ++i) { a[i] = p0[i]; a[8 + i] = p1[i]; }
        }
#pragma unroll
        for (int t = 0; t < 2; ++t) {
            const int col = wn * 32 + t * 16 + (lane & 15);
            const half_t* p0 = &lb[col * 40 + kh];
            const half_t* p1 = &lb[col * 40 + 16 + kh];
            v16h b;
#pragma unroll
            for (int i = 0; i < 8; ++i) { b[i] = p0[i]; b[8 + i] = p1[i]; }
            if (t == 0)
                acc0 = __builtin_amdgcn_wmma_f32_16x16x32_f16(false, a, false, b,
                                                              (short)0, acc0, false, false);
            else
                acc1 = __builtin_amdgcn_wmma_f32_16x16x32_f16(false, a, false, b,
                                                              (short)0, acc1, false, false);
        }
    }

    const int mrow = m_blk + wm * 16 + (lane >> 4) * 8;
#pragma unroll
    for (int t = 0; t < 2; ++t) {
        const int ncol = n_blk + wn * 32 + t * 16 + (lane & 15);
        const float bv = bias[ncol];
#pragma unroll
        for (int i = 0; i < 8; ++i) {
            float x = (t == 0 ? acc0[i] : acc1[i]) + bv;
            if (act == 1)      x = fmaxf(x, 0.0f);
            else if (act == 2) x = tanhf(x);
            C[(size_t)(mrow + i) * N + ncol] = x;
        }
    }
}

// ---------------------------------------------------------------------------
// Fused v-region partial scores:
//   e = relu(v_tile @ W_tile + vbias);  partial = sum_n tanh(e + h_a)*alpha_w
//   atomicAdd into scores[b, r]  (softmax is shift-invariant -> alpha_b dropped)
// Block tile 128(M) x 128(N); 8 waves as 4(M) x 2(N); wave = 32x64 = 8 WMMA
// accum tiles (64 VGPRs).  Grid (N/128, M/128) with n fastest so the 4
// N-blocks sharing an M-tile of v run adjacently and hit L2 on v rows.
// Weight L2 traffic: (50176/128) x 4MB ~ 1.6 GB.
// ---------------------------------------------------------------------------
__global__ __launch_bounds__(256)
void vscores_kernel(const float* __restrict__ V,      // [B*R, D]
                    const float* __restrict__ W,      // [D, H]
                    const float* __restrict__ vbias,  // [H]
                    const float* __restrict__ h_a,    // [B, H]
                    const float* __restrict__ aw,     // [H]
                    float* __restrict__ scores)       // [B, NREG], pre-zeroed
{
    __shared__ half_t la[128 * 32];   // 8 KB
    __shared__ half_t lb[128 * 40];   // 10 KB
    const int tid  = threadIdx.x;
    const int lane = tid & 31;
    const int wave = tid >> 5;
    const int wm   = wave >> 1;       // 0..3
    const int wn   = wave & 1;        // 0..1
    const int row0  = blockIdx.y * 128;
    const int n_blk = blockIdx.x * 128;

    v8f acc[2][4] = {};

    for (int k0 = 0; k0 < D_IN; k0 += 32) {
        __syncthreads();
        // A tile 128x32: 4 passes, one float4/thread, packed b64 store
#pragma unroll
        for (int p = 0; p < 4; ++p) {
            const int ml = (tid >> 3) + p * 32;
            const int kl = (tid & 7) << 2;
            float4 f = *(const float4*)(V + (size_t)(row0 + ml) * D_IN + k0 + kl);
            *(uint2*)(&la[ml * 32 + kl]) = make_uint2(pk(f.x, f.y), pk(f.z, f.w));
        }
        // B tile 32x128 transposed, k-pair packed b32 stores
#pragma unroll
        for (int p = 0; p < 2; ++p) {
            const int k  = ((tid >> 5) + p * 8) << 1;
            const int n4 = (tid & 31) << 2;
            float4 f0 = *(const float4*)(W + (size_t)(k0 + k)     * H_DIM + n_blk + n4);
            float4 f1 = *(const float4*)(W + (size_t)(k0 + k + 1) * H_DIM + n_blk + n4);
            *(unsigned*)(&lb[(n4 + 0) * 40 + k]) = pk(f0.x, f1.x);
            *(unsigned*)(&lb[(n4 + 1) * 40 + k]) = pk(f0.y, f1.y);
            *(unsigned*)(&lb[(n4 + 2) * 40 + k]) = pk(f0.z, f1.z);
            *(unsigned*)(&lb[(n4 + 3) * 40 + k]) = pk(f0.w, f1.w);
        }
        __syncthreads();

        const int kh = (lane >> 4) << 3;
        v16h a[2];
#pragma unroll
        for (int mi = 0; mi < 2; ++mi) {
            const int row = wm * 32 + mi * 16 + (lane & 15);
            const half_t* p0 = &la[row * 32 + kh];
            const half_t* p1 = &la[row * 32 + 16 + kh];
#pragma unroll
            for (int i = 0; i < 8; ++i) { a[mi][i] = p0[i]; a[mi][8 + i] = p1[i]; }
        }
#pragma unroll
        for (int ni = 0; ni < 4; ++ni) {
            const int col = wn * 64 + ni * 16 + (lane & 15);
            const half_t* p0 = &lb[col * 40 + kh];
            const half_t* p1 = &lb[col * 40 + 16 + kh];
            v16h b;
#pragma unroll
            for (int i = 0; i < 8; ++i) { b[i] = p0[i]; b[8 + i] = p1[i]; }
#pragma unroll
            for (int mi = 0; mi < 2; ++mi)
                acc[mi][ni] = __builtin_amdgcn_wmma_f32_16x16x32_f16(
                    false, a[mi], false, b, (short)0, acc[mi][ni], false, false);
        }
    }

    // fused epilogue: relu, +h_a, tanh, *alpha_w, reduce over this N-slab
    const int mbase = (lane >> 4) << 3;               // rows 0-7 or 8-15 of tile
#pragma unroll
    for (int mi = 0; mi < 2; ++mi) {
        float psum[8] = {0.f, 0.f, 0.f, 0.f, 0.f, 0.f, 0.f, 0.f};
#pragma unroll
        for (int ni = 0; ni < 4; ++ni) {
            const int n   = n_blk + wn * 64 + ni * 16 + (lane & 15);
            const float bv  = vbias[n];
            const float awn = aw[n];
#pragma unroll
            for (int i = 0; i < 8; ++i) {
                const int mg = row0 + wm * 32 + mi * 16 + mbase + i;
                const int bb = mg / R_SZ;
                float e = fmaxf(acc[mi][ni][i] + bv, 0.0f);       // relu (v_embed)
                psum[i] += tanhf(e + h_a[(size_t)bb * H_DIM + n]) * awn;
            }
        }
        // butterfly across 16 lanes of each half (halves hold different rows)
#pragma unroll
        for (int off = 8; off >= 1; off >>= 1) {
#pragma unroll
            for (int i = 0; i < 8; ++i)
                psum[i] += __shfl_xor(psum[i], off, 32);
        }
        if ((lane & 15) == 0) {
#pragma unroll
            for (int i = 0; i < 8; ++i) {
                const int mg = row0 + wm * 32 + mi * 16 + mbase + i;
                const int bb = mg / R_SZ, rr = mg % R_SZ;
                atomicAdd(&scores[(size_t)bb * NREG + rr], psum[i]);
            }
        }
    }
}

__global__ __launch_bounds__(256)
void zero_kernel(float* __restrict__ p, int n)
{
    const int i = blockIdx.x * 256 + threadIdx.x;
    if (i < n) p[i] = 0.0f;
}

// score for the appended s-region (s_a used directly, NO relu; alpha_b dropped)
__global__ __launch_bounds__(128)
void srow_score_kernel(const float* __restrict__ s_a, const float* __restrict__ h_a,
                       const float* __restrict__ aw, float* __restrict__ scores)
{
    __shared__ float red[128];
    const int b = blockIdx.x, tid = threadIdx.x;
    float p = 0.0f;
    for (int hh = tid; hh < H_DIM; hh += 128)
        p += tanhf(s_a[(size_t)b * H_DIM + hh] + h_a[(size_t)b * H_DIM + hh]) * aw[hh];
    red[tid] = p;
    __syncthreads();
    for (int s = 64; s > 0; s >>= 1) {
        if (tid < s) red[tid] += red[tid + s];
        __syncthreads();
    }
    if (tid == 0) scores[(size_t)b * NREG + R_SZ] = red[0];
}

// per-batch softmax over 197 regions
__global__ __launch_bounds__(256)
void softmax_kernel(const float* __restrict__ scores, float* __restrict__ alpha)
{
    __shared__ float red[256];
    const int b = blockIdx.x, tid = threadIdx.x;
    float x = (tid < NREG) ? scores[(size_t)b * NREG + tid] : -3.4e38f;
    red[tid] = x;
    __syncthreads();
    for (int s = 128; s > 0; s >>= 1) {
        if (tid < s) red[tid] = fmaxf(red[tid], red[tid + s]);
        __syncthreads();
    }
    const float mx = red[0];
    __syncthreads();
    const float e = (tid < NREG) ? __expf(x - mx) : 0.0f;
    red[tid] = e;
    __syncthreads();
    for (int s = 128; s > 0; s >>= 1) {
        if (tid < s) red[tid] += red[tid + s];
        __syncthreads();
    }
    const float inv = 1.0f / red[0];
    if (tid < NREG) alpha[(size_t)b * NREG + tid] = e * inv;
}

// tmp[b,d] = sum_r alpha[b,r]*v[b,r,d] + alpha[b,196]*s_p[b,d] + h_p[b,d]
// second (final) streaming pass over v; prefetch ahead 8 rows.
__global__ __launch_bounds__(256)
void context_kernel(const float* __restrict__ V, const float* __restrict__ s_p,
                    const float* __restrict__ h_p, const float* __restrict__ alpha,
                    float* __restrict__ tmp)
{
    __shared__ float sal[NREG];
    const int b    = blockIdx.x >> 1;
    const int half = blockIdx.x & 1;
    const int d4   = half * 256 + threadIdx.x;     // float4 index, 0..511
    for (int i = threadIdx.x; i < NREG; i += 256)
        sal[i] = alpha[(size_t)b * NREG + i];
    __syncthreads();

    const float4* v4 = (const float4*)(V + (size_t)b * R_SZ * D_IN);
    float4 c = make_float4(0.f, 0.f, 0.f, 0.f);
    for (int r = 0; r < R_SZ; ++r) {
        if (r + 8 < R_SZ)
            __builtin_prefetch(&v4[(size_t)(r + 8) * (D_IN / 4) + d4], 0, 1);
        const float a = sal[r];
        const float4 x = v4[(size_t)r * (D_IN / 4) + d4];
        c.x += a * x.x; c.y += a * x.y; c.z += a * x.z; c.w += a * x.w;
    }
    {
        const float a = sal[R_SZ];
        const float4 x = ((const float4*)(s_p + (size_t)b * D_IN))[d4];
        c.x += a * x.x; c.y += a * x.y; c.z += a * x.z; c.w += a * x.w;
    }
    {
        const float4 x = ((const float4*)(h_p + (size_t)b * D_IN))[d4];
        c.x += x.x; c.y += x.y; c.z += x.z; c.w += x.w;
    }
    ((float4*)(tmp + (size_t)b * D_IN))[d4] = c;
}

// ---------------------------------------------------------------------------
extern "C" void kernel_launch(void* const* d_in, const int* in_sizes, int n_in,
                              void* d_out, int out_size, void* d_ws, size_t ws_size,
                              hipStream_t stream)
{
    const float* v        = (const float*)d_in[0];   // [B,R,D]
    const float* s        = (const float*)d_in[1];   // [B,D]
    const float* h        = (const float*)d_in[2];   // [B,D]
    const float* v_att_w  = (const float*)d_in[3];   // [D,H]
    const float* v_att_b  = (const float*)d_in[4];   // [H]
    const float* s_proj_w = (const float*)d_in[5];   // [D,D]
    const float* s_proj_b = (const float*)d_in[6];   // [D]
    const float* s_att_w  = (const float*)d_in[7];   // [D,H]
    const float* s_att_b  = (const float*)d_in[8];   // [H]
    const float* h_proj_w = (const float*)d_in[9];   // [D,D]
    const float* h_proj_b = (const float*)d_in[10];  // [D]
    const float* h_att_w  = (const float*)d_in[11];  // [D,H]
    const float* h_att_b  = (const float*)d_in[12];  // [H]
    const float* alpha_w  = (const float*)d_in[13];  // [H,1] -> flat [H]
    // d_in[14] = alpha_b: constant shift, cancels in softmax
    const float* ctx_w    = (const float*)d_in[15];  // [D,D]
    const float* ctx_b    = (const float*)d_in[16];  // [D]
    float* out = (float*)d_out;                      // [B,D]

    // workspace layout (floats); total ~7.8 MB
    float* ws      = (float*)d_ws;
    float* s_p     = ws;                             // 256*2048
    float* h_p     = s_p + (size_t)B_SZ * D_IN;      // 256*2048
    float* s_a     = h_p + (size_t)B_SZ * D_IN;      // 256*512
    float* h_a     = s_a + (size_t)B_SZ * H_DIM;     // 256*512
    float* scores  = h_a + (size_t)B_SZ * H_DIM;     // 256*197
    float* alpha   = scores + (size_t)B_SZ * NREG;   // 256*197
    float* tmp     = alpha + (size_t)B_SZ * NREG;    // 256*2048

    const dim3 blk(256);

    zero_kernel<<<dim3((B_SZ * NREG + 255) / 256), blk, 0, stream>>>(scores, B_SZ * NREG);

    // s_p = relu(s @ s_proj_w + b)
    gemm_bias_act_kernel<<<dim3(D_IN / 128, B_SZ / 32), blk, 0, stream>>>(
        s, s_proj_w, s_proj_b, s_p, B_SZ, D_IN, D_IN, 1);
    // h_p = tanh(h @ h_proj_w + b)
    gemm_bias_act_kernel<<<dim3(D_IN / 128, B_SZ / 32), blk, 0, stream>>>(
        h, h_proj_w, h_proj_b, h_p, B_SZ, D_IN, D_IN, 2);
    // s_a = s_p @ s_att_w + b
    gemm_bias_act_kernel<<<dim3(H_DIM / 128, B_SZ / 32), blk, 0, stream>>>(
        s_p, s_att_w, s_att_b, s_a, B_SZ, H_DIM, D_IN, 0);
    // h_a = h_p @ h_att_w + b
    gemm_bias_act_kernel<<<dim3(H_DIM / 128, B_SZ / 32), blk, 0, stream>>>(
        h_p, h_att_w, h_att_b, h_a, B_SZ, H_DIM, D_IN, 0);

    // fused big GEMM + score epilogue; n fastest for L2 reuse of v rows
    vscores_kernel<<<dim3(H_DIM / 128, (B_SZ * R_SZ) / 128), blk, 0, stream>>>(
        v, v_att_w, v_att_b, h_a, alpha_w, scores);
    srow_score_kernel<<<dim3(B_SZ), dim3(128), 0, stream>>>(s_a, h_a, alpha_w, scores);
    softmax_kernel<<<dim3(B_SZ), blk, 0, stream>>>(scores, alpha);
    context_kernel<<<dim3(B_SZ * 2), blk, 0, stream>>>(v, s_p, h_p, alpha, tmp);
    // z_t = tanh((context + h_p) @ ctx_w + b)
    gemm_bias_act_kernel<<<dim3(D_IN / 128, B_SZ / 32), blk, 0, stream>>>(
        tmp, ctx_w, ctx_b, out, B_SZ, D_IN, D_IN, 2);

    (void)in_sizes; (void)n_in; (void)out_size; (void)ws_size;
}
